// GreedyMatcher_1228360646759
// MI455X (gfx1250) — matrix operand order repaired and checked by reading, not compiled, not run on hardware
//
#include <hip/hip_runtime.h>

// ---------------------------------------------------------------------------
// PDHG LP solver for MI455X (gfx1250, wave32, WMMA).
//   x_new = max(x + tau*(1 - y@S), 0)            GEMM1: [512,256]x[256,4096]
//   y_new = max(y + sig*((2x_new - x)@S^T - b),0) GEMM2: [512,4096]x[4096,256]
// S is binary -> exact in bf16. Float operands split hi/lo bf16 (products with
// {0,1} are exact), accumulated in f32 via v_wmma_f32_16x16x32_bf16 (2 per k).
// Operands are stored pre-packed in WMMA fragment layout so the GEMM inner
// loop is pure {3x 32B loads + 2x WMMA}.
// ---------------------------------------------------------------------------

typedef __attribute__((ext_vector_type(16))) __bf16 v16bf;
typedef __attribute__((ext_vector_type(8)))  float  v8f;

#define B_DIM 512
#define M_DIM 256
#define N_DIM 4096

__device__ __forceinline__ __bf16 f32_to_bf16(float f) {
  unsigned u = __builtin_bit_cast(unsigned, f);
  u += 0x7FFFu + ((u >> 16) & 1u);              // round-to-nearest-even
  unsigned short s = (unsigned short)(u >> 16);
  return __builtin_bit_cast(__bf16, s);
}
__device__ __forceinline__ float bf16_to_f32(__bf16 h) {
  unsigned short s = __builtin_bit_cast(unsigned short, h);
  unsigned u = ((unsigned)s) << 16;
  return __builtin_bit_cast(float, u);
}

// --------------------------------------------------------------------------
// Pack S (B-operand of GEMM1) and S^T (B-operand of GEMM2) into wave32 WMMA
// B-fragment layout: lanes 0-15 hold K={0-7,16-23} (col N=lane), lanes 16-31 hold
// K={8-15,24-31} (col N=lane-16); 16 bf16 contiguous per lane.
// Sfrag : [kt=0..7][nt=0..255][lane][e]   (K over m=256,  N over n=4096)
// STfrag: [kt=0..127][nt=0..15][lane][e]  (K over n=4096, N over m=256)
// --------------------------------------------------------------------------
__global__ void __launch_bounds__(256) prep_frags(const float* __restrict__ S,
                                                  __bf16* __restrict__ Sfrag,
                                                  __bf16* __restrict__ STfrag) {
  const size_t gid = (size_t)blockIdx.x * 256 + threadIdx.x;
  const size_t n1 = (size_t)8 * 256 * 512;   // 1,048,576 entries
  if (gid < n1) {
    const size_t t = gid >> 9;
    const int rem = (int)(gid & 511);
    const int lane = rem >> 4, e = rem & 15;
    const int kt = (int)(t >> 8), nt = (int)(t & 255);
    const int K = (e & 7) + ((lane >> 4) << 3) + ((e >> 3) << 4);
    const int N = lane & 15;
    Sfrag[gid] = f32_to_bf16(S[(size_t)(kt * 32 + K) * N_DIM + nt * 16 + N]);
  } else {
    const size_t g2 = gid - n1;
    const size_t t = g2 >> 9;
    const int rem = (int)(g2 & 511);
    const int lane = rem >> 4, e = rem & 15;
    const int kt = (int)(t >> 4), nt = (int)(t & 15);
    const int K = (e & 7) + ((lane >> 4) << 3) + ((e >> 3) << 4);
    const int N = lane & 15;
    // S^T[K, N] == S[N_m, K_n]
    STfrag[g2] = f32_to_bf16(S[(size_t)(nt * 16 + N) * N_DIM + (kt * 32 + K)]);
  }
}

// --------------------------------------------------------------------------
// Power iteration (single WGP): v <- S^T(Sv)/||.|| x30, then tau = 0.9/||Sv||.
// ~63 MFLOP total: negligible.
// --------------------------------------------------------------------------
__global__ void __launch_bounds__(1024) power_iter(const float* __restrict__ S,
                                                   float* __restrict__ tau_out) {
  __shared__ float v[N_DIM];
  __shared__ float t[M_DIM];
  __shared__ float red[1024];
  const int tid = threadIdx.x;
  for (int i = tid; i < N_DIM; i += 1024) v[i] = 1.0f;
  __syncthreads();
  const int r = tid >> 2, q = tid & 3;
  for (int it = 0; it < 30; ++it) {
    // t = S @ v  (4 threads per row)
    float p = 0.0f;
    {
      const float* Sr = S + (size_t)r * N_DIM + q * 1024;
      const float* vq = v + q * 1024;
      for (int j = 0; j < 1024; ++j) p += Sr[j] * vq[j];
    }
    red[tid] = p;
    __syncthreads();
    if (q == 0) t[r] = red[tid] + red[tid + 1] + red[tid + 2] + red[tid + 3];
    __syncthreads();
    // u = S^T @ t (4 cols per thread, coalesced across tid), norm^2
    float u[4];
    float nrm = 0.0f;
    for (int c4 = 0; c4 < 4; ++c4) {
      const int c = tid + c4 * 1024;
      float s = 0.0f;
      for (int rr = 0; rr < M_DIM; ++rr) s += S[(size_t)rr * N_DIM + c] * t[rr];
      u[c4] = s;
      nrm += s * s;
    }
    red[tid] = nrm;
    __syncthreads();
    for (int off = 512; off > 0; off >>= 1) {
      if (tid < off) red[tid] += red[tid + off];
      __syncthreads();
    }
    const float inv = 1.0f / sqrtf(red[0]);
    for (int c4 = 0; c4 < 4; ++c4) v[tid + c4 * 1024] = u[c4] * inv;
    __syncthreads();
  }
  // L = ||S v||  (v normalized -> v^T S^T S v = ||Sv||^2)
  float p = 0.0f;
  {
    const float* Sr = S + (size_t)r * N_DIM + q * 1024;
    const float* vq = v + q * 1024;
    for (int j = 0; j < 1024; ++j) p += Sr[j] * vq[j];
  }
  red[tid] = p;
  __syncthreads();
  if (q == 0) t[r] = red[tid] + red[tid + 1] + red[tid + 2] + red[tid + 3];
  __syncthreads();
  if (tid < 256) red[tid] = t[tid] * t[tid];
  __syncthreads();
  for (int off = 128; off > 0; off >>= 1) {
    if (tid < off) red[tid] += red[tid + off];
    __syncthreads();
  }
  if (tid == 0) tau_out[0] = 0.9f / sqrtf(red[0]);
}

__global__ void __launch_bounds__(256) zero_f32(float* __restrict__ p, int n) {
  const int i = blockIdx.x * 256 + threadIdx.x;
  if (i < n) p[i] = 0.0f;
}

// --------------------------------------------------------------------------
// GEMM1 + x epilogue. One 16x16 output tile per wave, K-loop = 256/32 = 8.
// A fragments (y hi/lo) come pre-packed from y_update's epilogue.
// Epilogue also emits z = 2*x_new - x_old pre-packed for GEMM2 (A layout:
// lane half 0 -> K {0-7,16-23}, half 1 -> K {8-15,24-31}, row M = lane&15).
// --------------------------------------------------------------------------
__global__ void __launch_bounds__(256) x_update(
    const __bf16* __restrict__ yfH, const __bf16* __restrict__ yfL,
    const __bf16* __restrict__ Sfrag,
    const float* __restrict__ x_src, float* __restrict__ x_dst,
    __bf16* __restrict__ zfH, __bf16* __restrict__ zfL,
    const float* __restrict__ taup) {
  const int wave = threadIdx.x >> 5;
  const int lane = threadIdx.x & 31;
  const int tile = blockIdx.x * 8 + wave;      // 0..8191
  const int trow = tile >> 8;                  // batch tile 0..31
  const int tcol = tile & 255;                 // n tile 0..255
  const int half = lane >> 4;
  const int rsub = lane & 15;
  const float tau = taup[0];

  v8f acc = {};
#pragma unroll
  for (int kt = 0; kt < 8; ++kt) {
    const size_t af = (((size_t)trow * 8 + kt) * 32 + lane) * 16;
    const v16bf ahi = *(const v16bf*)(yfH + af);
    const v16bf alo = *(const v16bf*)(yfL + af);
    const size_t bf = (((size_t)kt * 256 + tcol) * 32 + lane) * 16;
    const v16bf bb = *(const v16bf*)(Sfrag + bf);
    if (kt < 7)
      __builtin_prefetch((const void*)(Sfrag + (((size_t)(kt + 1) * 256 + tcol) * 32 + lane) * 16), 0, 1);
    acc = __builtin_amdgcn_wmma_f32_16x16x32_bf16(false, ahi, false, bb,
                                                  (short)0, acc, false, false);
    acc = __builtin_amdgcn_wmma_f32_16x16x32_bf16(false, alo, false, bb,
                                                  (short)0, acc, false, false);
  }

  // D layout: col N = lane&15; VGPR r -> row M = r + half*8
  const int col = tcol * 16 + rsub;
  const int kt2 = col >> 5;
  const int k = col & 31;
  const int grp = k >> 3;
  const int halfA = grp & 1;
  const int eA = (k & 7) + ((grp & 2) << 2);
#pragma unroll
  for (int r = 0; r < 8; ++r) {
    const int row = trow * 16 + r + half * 8;
    const size_t idx = (size_t)row * N_DIM + col;
    const float xv = x_src[idx];
    float xn = xv + tau * (1.0f - acc[r]);
    xn = xn > 0.0f ? xn : 0.0f;
    x_dst[idx] = xn;
    const float z = 2.0f * xn - xv;
    const __bf16 h = f32_to_bf16(z);
    const __bf16 l = f32_to_bf16(z - bf16_to_f32(h));
    const int laneA = halfA * 16 + (r + half * 8);
    const size_t fidx = (((size_t)trow * 128 + kt2) * 32 + laneA) * 16 + eA;
    zfH[fidx] = h;
    zfL[fidx] = l;
  }
}

// --------------------------------------------------------------------------
// GEMM2 + y epilogue. K = 4096 split 4-ways (4 waves/tile, 32 k-steps each),
// partials reduced in LDS (fixed order -> deterministic). Epilogue updates y
// in place and emits y hi/lo pre-packed for next iteration's GEMM1.
// --------------------------------------------------------------------------
__global__ void __launch_bounds__(256) y_update(
    const __bf16* __restrict__ zfH, const __bf16* __restrict__ zfL,
    const __bf16* __restrict__ STfrag,
    const float* __restrict__ bvec, float* __restrict__ y,
    __bf16* __restrict__ yfH, __bf16* __restrict__ yfL,
    const float* __restrict__ taup) {
  __shared__ float red[8][8][32];
  const int wave = threadIdx.x >> 5;
  const int lane = threadIdx.x & 31;
  const int wt = wave >> 2;                 // tile within block (0..1)
  const int kq = wave & 3;                  // K quadrant
  const int tile = blockIdx.x * 2 + wt;     // 0..511
  const int trow = tile >> 4;               // batch tile 0..31
  const int tcol = tile & 15;               // m tile 0..15
  const int half = lane >> 4;
  const int rsub = lane & 15;
  const float sig = taup[0];

  v8f acc = {};
  for (int kt = kq * 32; kt < kq * 32 + 32; ++kt) {
    const size_t af = (((size_t)trow * 128 + kt) * 32 + lane) * 16;
    const v16bf ahi = *(const v16bf*)(zfH + af);
    const v16bf alo = *(const v16bf*)(zfL + af);
    const size_t bf = (((size_t)kt * 16 + tcol) * 32 + lane) * 16;
    const v16bf bb = *(const v16bf*)(STfrag + bf);
    __builtin_prefetch((const void*)(STfrag + (((size_t)(kt + 1) * 16 + tcol) * 32 + lane) * 16), 0, 1);
    acc = __builtin_amdgcn_wmma_f32_16x16x32_bf16(false, ahi, false, bb,
                                                  (short)0, acc, false, false);
    acc = __builtin_amdgcn_wmma_f32_16x16x32_bf16(false, alo, false, bb,
                                                  (short)0, acc, false, false);
  }
#pragma unroll
  for (int r = 0; r < 8; ++r) red[wave][r][lane] = acc[r];
  __syncthreads();
  if (kq == 0) {
    const int w0 = wt * 4;
    const int col = tcol * 16 + rsub;
    const int kt2 = col >> 5;
    const int k = col & 31;
    const int grp = k >> 3;
    const int halfA = grp & 1;
    const int eA = (k & 7) + ((grp & 2) << 2);
#pragma unroll
    for (int r = 0; r < 8; ++r) {
      const float s = red[w0][r][lane] + red[w0 + 1][r][lane] +
                      red[w0 + 2][r][lane] + red[w0 + 3][r][lane];
      const int row = trow * 16 + r + half * 8;
      const size_t idx = (size_t)row * M_DIM + col;
      float t = y[idx] + sig * (s - bvec[idx]);
      t = t > 0.0f ? t : 0.0f;
      y[idx] = t;
      const __bf16 h = f32_to_bf16(t);
      const __bf16 l = f32_to_bf16(t - bf16_to_f32(h));
      const int laneA = halfA * 16 + (r + half * 8);
      const size_t fidx = (((size_t)trow * 8 + kt2) * 32 + laneA) * 16 + eA;
      yfH[fidx] = h;
      yfL[fidx] = l;
    }
  }
}

// --------------------------------------------------------------------------
extern "C" void kernel_launch(void* const* d_in, const int* in_sizes, int n_in,
                              void* d_out, int out_size, void* d_ws, size_t ws_size,
                              hipStream_t stream) {
  const float* X = (const float*)d_in[0];   // [512,16,16] == b [512,256]
  const float* S = (const float*)d_in[1];   // [256,4096]
  char* ws = (char*)d_ws;
  const size_t MB = 1024 * 1024;
  // workspace layout (all 256B aligned), ~29 MB total:
  float*  tau    = (float*)(ws);                                  // 256 B
  __bf16* Sfrag  = (__bf16*)(ws + 256);                           // 2 MB
  __bf16* STfrag = (__bf16*)(ws + 256 + 2 * MB);                  // 2 MB
  float*  x0     = (float*)(ws + 256 + 4 * MB);                   // 8 MB
  float*  y      = (float*)(ws + 256 + 12 * MB);                  // 512 KB
  __bf16* yfH    = (__bf16*)(ws + 256 + 12 * MB + 512 * 1024);    // 256 KB
  __bf16* yfL    = (__bf16*)(ws + 256 + 12 * MB + 768 * 1024);    // 256 KB
  float*  x1     = (float*)(ws + 256 + 13 * MB);                  // 8 MB
  __bf16* zfH    = (__bf16*)(ws + 256 + 21 * MB);                 // 4 MB
  __bf16* zfL    = (__bf16*)(ws + 256 + 25 * MB);                 // 4 MB

  prep_frags<<<8192, 256, 0, stream>>>(S, Sfrag, STfrag);
  power_iter<<<1, 1024, 0, stream>>>(S, tau);
  // zero x0 + y + yfH + yfL (contiguous region): 8MB + 512KB + 512KB
  zero_f32<<<(2359296 + 255) / 256, 256, 0, stream>>>(x0, 2359296);

  for (int it = 0; it < 200; ++it) {
    float* xs = (it & 1) ? x1 : x0;
    float* xd = (it & 1) ? x0 : x1;
    x_update<<<1024, 256, 0, stream>>>(yfH, yfL, Sfrag, xs, xd, zfH, zfL, tau);
    y_update<<<256, 256, 0, stream>>>(zfH, zfL, STfrag, X, y, yfH, yfL, tau);
  }
  // after it=199 (odd), final x is in x0
  hipMemcpyAsync(d_out, x0, (size_t)B_DIM * N_DIM * sizeof(float),
                 hipMemcpyDeviceToDevice, stream);
}
